// Critic_26938034880700
// MI455X (gfx1250) — compile-verified, matrix operand order checked by reading, not run on hardware
//
#include <hip/hip_runtime.h>
#include <hip/hip_bf16.h>
#include <float.h>

// ---------------------------------------------------------------------------
// Critic GNN, fused for MI455X (gfx1250, wave32):
//   per-edge MLP (10->32 relu, 32->32) done 16 edges/wave via
//   v_wmma_f32_16x16x32_bf16 (bf16 in, f32 accumulate), segment_sum via
//   L2-resident global_atomic_add_f32, then two small reduction kernels.
//   ReLU via v_med3_f32; next-tile streams hidden with global_prefetch_b8.
// ---------------------------------------------------------------------------

#define N_HID 32

typedef __attribute__((ext_vector_type(16))) __bf16 v16bf;
typedef __attribute__((ext_vector_type(8)))  __bf16 v8bf;
typedef __attribute__((ext_vector_type(8)))  float  v8f;

__device__ __forceinline__ v8f wmma_bf16(v16bf a, v16bf b, v8f c) {
    return __builtin_amdgcn_wmma_f32_16x16x32_bf16(
        /*neg_a=*/false, a, /*neg_b=*/false, b,
        /*c_mod=*/(short)0, c, /*reuse_a=*/false, /*reuse_b=*/false);
}

// Single-VALU ReLU: med3(x, 0, +MAX) clamps below at 0 (WMMA outputs are canonical).
__device__ __forceinline__ float relu1(float x) {
    return __builtin_amdgcn_fmed3f(x, 0.0f, FLT_MAX);
}

__global__ void __launch_bounds__(256)
zero_kernel(float* __restrict__ p, int n) {
    int i = blockIdx.x * blockDim.x + threadIdx.x;
    if (i < n) p[i] = 0.0f;
}

// One wave processes 16 edges at a time:
//   A  (16x32 bf16) = [x_i(4) | x_j(4) | edge_attr(2) | zero-pad to 32]
//   h  = relu(A @ W1 + b1)        -> 2x v_wmma_f32_16x16x32_bf16
//   msg= h @ W2 + b2              -> 2x v_wmma_f32_16x16x32_bf16
//   agg[row[e]] += msg[e]         -> global_atomic_add_f32 (L2 resident)
__global__ void __launch_bounds__(256)
edge_kernel(const float* __restrict__ x,
            const int*   __restrict__ ei,
            const float* __restrict__ ea,
            const float* __restrict__ W1,
            const float* __restrict__ b1,
            const float* __restrict__ W2,
            const float* __restrict__ b2,
            float* __restrict__ agg,
            int nEdges) {
    constexpr int WAVES = 8;
    __shared__ __bf16 hbuf[WAVES][16 * N_HID];   // 1KB per wave h-tile staging

    const int lane = threadIdx.x & 31;
    const int wid  = threadIdx.x >> 5;
    const int half = lane >> 4;     // K half select (A/B layout)
    const int nl   = lane & 15;     // matrix row (A) / col (B,C)

    // ---- Preload B operands in WMMA B layout (K=32 x N=16 per reg set) ----
    // v[j] holds element (k, n) with k = 16*(j/8) + 8*half + (j%8), n = nl.
    v16bf B1lo, B1hi, B2lo, B2hi;
    #pragma unroll
    for (int j = 0; j < 16; ++j) {
        int k = ((j >> 3) << 4) + (half << 3) + (j & 7);
        float w1l = (k < 10) ? W1[k * N_HID + nl]      : 0.0f;  // K padded 10->32
        float w1h = (k < 10) ? W1[k * N_HID + 16 + nl] : 0.0f;
        B1lo[j] = (__bf16)w1l;
        B1hi[j] = (__bf16)w1h;
        B2lo[j] = (__bf16)W2[k * N_HID + nl];
        B2hi[j] = (__bf16)W2[k * N_HID + 16 + nl];
    }
    const float bias1lo = b1[nl], bias1hi = b1[16 + nl];
    const float bias2lo = b2[nl], bias2hi = b2[16 + nl];

    const int nTiles = (nEdges + 15) >> 4;
    const int gw     = blockIdx.x * WAVES + wid;
    const int stride = gridDim.x * WAVES;

    for (int tile = gw; tile < nTiles; tile += stride) {
        const int e      = (tile << 4) + nl;
        const bool valid = (e < nEdges);
        const int row  = valid ? ei[e]          : 0;
        const int colI = valid ? ei[nEdges + e] : 0;

        // Prefetch next tile's streaming data (edge ids + attrs). These are
        // the only true HBM streams; the wave's next tile is `stride` tiles
        // away so the HW streamer won't catch it. global_prefetch_b8 is
        // counter-free.
        {
            const int pe = ((tile + stride) << 4) + nl;
            if (pe < nEdges) {
                __builtin_prefetch(ei + pe, 0, 3);
                __builtin_prefetch(ei + nEdges + pe, 0, 3);
                __builtin_prefetch(ea + 2 * pe, 0, 3);
            }
        }

        // ---- Build A (16x32 bf16) in WMMA A layout -------------------------
        // lane<16 owns K 0..7 (x_i|x_j) and K16..23 (pad); lane>=16 owns
        // K 8..15 (edge_attr|pad) and K24..31 (pad).
        v16bf A = {};
        if (valid) {
            if (half == 0) {
                const float4 xi = *(const float4*)(x + 4 * row);
                const float4 xj = *(const float4*)(x + 4 * colI);
                A[0] = (__bf16)xi.x; A[1] = (__bf16)xi.y;
                A[2] = (__bf16)xi.z; A[3] = (__bf16)xi.w;
                A[4] = (__bf16)xj.x; A[5] = (__bf16)xj.y;
                A[6] = (__bf16)xj.z; A[7] = (__bf16)xj.w;
            } else {
                const float2 eav = *(const float2*)(ea + 2 * e);
                A[0] = (__bf16)eav.x; A[1] = (__bf16)eav.y;
            }
        }

        // ---- GEMM1: h = relu(A @ W1 + b1), bias preloaded into C ----------
        v8f c1lo, c1hi;
        #pragma unroll
        for (int i = 0; i < 8; ++i) { c1lo[i] = bias1lo; c1hi[i] = bias1hi; }
        c1lo = wmma_bf16(A, B1lo, c1lo);
        c1hi = wmma_bf16(A, B1hi, c1hi);

        // ---- ReLU + C-layout -> A-layout relayout via per-wave LDS --------
        __bf16* hb = &hbuf[wid][0];
        #pragma unroll
        for (int r = 0; r < 8; ++r) {
            int m = r + (half << 3);
            hb[m * N_HID + nl]      = (__bf16)relu1(c1lo[r]);
            hb[m * N_HID + 16 + nl] = (__bf16)relu1(c1hi[r]);
        }
        // Intra-wave LDS RAW: LDS is in-order per wave; wait + pin ordering.
        asm volatile("s_wait_dscnt 0" ::: "memory");

        const v8bf* plo = (const v8bf*)(hb + nl * N_HID + (half << 3));
        const v8bf* phi = (const v8bf*)(hb + nl * N_HID + 16 + (half << 3));
        v8bf alo = *plo;
        v8bf ahi = *phi;
        asm volatile("" ::: "memory");   // keep these loads before next-iter stores

        v16bf A2;
        #pragma unroll
        for (int i = 0; i < 8; ++i) { A2[i] = alo[i]; A2[i + 8] = ahi[i]; }

        // ---- GEMM2: msg = h @ W2 + b2 -------------------------------------
        v8f c2lo, c2hi;
        #pragma unroll
        for (int i = 0; i < 8; ++i) { c2lo[i] = bias2lo; c2hi[i] = bias2hi; }
        c2lo = wmma_bf16(A2, B2lo, c2lo);
        c2hi = wmma_bf16(A2, B2hi, c2hi);

        // ---- segment_sum scatter: agg[row] += msg -------------------------
        // C layout: VGPR r holds edge (r + 8*half) of the tile, column nl(+16).
        #pragma unroll
        for (int r = 0; r < 8; ++r) {
            int et   = r + (half << 3);                 // edge within tile
            int orow = __shfl(row, et, 32);             // wave32 cross-lane
            if ((tile << 4) + et < nEdges) {
                atomicAdd(agg + (size_t)orow * N_HID + nl,      c2lo[r]);
                atomicAdd(agg + (size_t)orow * N_HID + 16 + nl, c2hi[r]);
            }
        }
    }
}

// Column-parallel pooling: pooled[0..3] = sum_n x[n][c], pooled[4..35] = sum_n agg[n][c-4]
__global__ void __launch_bounds__(256)
reduce_kernel(const float* __restrict__ x,
              const float* __restrict__ agg,
              float* __restrict__ pooled,
              int nNodes, int nChunks) {
    const int lane = threadIdx.x & 31;
    const int wid  = threadIdx.x >> 5;
    const int gw   = blockIdx.x * (blockDim.x >> 5) + wid;
    const int col   = gw % 36;
    const int chunk = gw / 36;
    if (chunk >= nChunks) return;

    float s = 0.0f;
    for (int n = chunk * 32 + lane; n < nNodes; n += nChunks * 32)
        s += (col < 4) ? x[(size_t)n * 4 + col]
                       : agg[(size_t)n * N_HID + (col - 4)];
    #pragma unroll
    for (int off = 16; off > 0; off >>= 1) s += __shfl_down(s, off, 32);
    if (lane == 0) atomicAdd(pooled + col, s);
}

__global__ void __launch_bounds__(64)
final_kernel(const float* __restrict__ pooled,
             const float* __restrict__ Wv,
             const float* __restrict__ bv,
             float* __restrict__ out) {
    __shared__ float red[64];
    int t = threadIdx.x;
    red[t] = (t < 36) ? pooled[t] * Wv[t] : 0.0f;
    __syncthreads();
    #pragma unroll
    for (int off = 32; off > 0; off >>= 1) {
        if (t < off) red[t] += red[t + off];
        __syncthreads();
    }
    if (t == 0) out[0] = red[0] + bv[0];
}

extern "C" void kernel_launch(void* const* d_in, const int* in_sizes, int n_in,
                              void* d_out, int out_size, void* d_ws, size_t ws_size,
                              hipStream_t stream) {
    const float* x  = (const float*)d_in[0];
    const int*   ei = (const int*)  d_in[1];
    const float* ea = (const float*)d_in[2];
    const float* W1 = (const float*)d_in[3];
    const float* b1 = (const float*)d_in[4];
    const float* W2 = (const float*)d_in[5];
    const float* b2 = (const float*)d_in[6];
    const float* Wv = (const float*)d_in[7];
    const float* bv = (const float*)d_in[8];
    float* out = (float*)d_out;

    const int nNodes = in_sizes[0] / 4;   // x is [N, 4]
    const int nEdges = in_sizes[1] / 2;   // edge_index is [2, E]

    // Workspace: agg [nNodes*32] f32, then pooled [36] f32.
    float* agg    = (float*)d_ws;
    float* pooled = agg + (size_t)nNodes * N_HID;
    const int zn  = nNodes * N_HID + 36;

    zero_kernel<<<(zn + 255) / 256, 256, 0, stream>>>(agg, zn);

    edge_kernel<<<1024, 256, 0, stream>>>(x, ei, ea, W1, b1, W2, b2, agg, nEdges);

    const int nChunks = 32;                       // 36 cols * 32 chunks = 1152 waves
    reduce_kernel<<<(36 * nChunks + 7) / 8, 256, 0, stream>>>(x, agg, pooled,
                                                              nNodes, nChunks);

    final_kernel<<<1, 64, 0, stream>>>(pooled, Wv, bv, out);
}
